// PersistenceWeightedPositionalEncoding_29841432773259
// MI455X (gfx1250) — matrix operand order repaired and checked by reading, not compiled, not run on hardware
//
#include <hip/hip_runtime.h>
#include <math.h>

#define BB   64
#define NN   262144
#define PP   256
#define DIM  256
#define D4   64
#define D2   128
#define BPN  (BB*PP)          // 16384 bins / rows

typedef __attribute__((ext_vector_type(2)))  float    v2f;
typedef __attribute__((ext_vector_type(4)))  float    v4f;
typedef __attribute__((ext_vector_type(8)))  float    v8f;
typedef __attribute__((ext_vector_type(4)))  int      v4i;
typedef __attribute__((ext_vector_type(16))) _Float16 v16h;

#if __has_builtin(__builtin_amdgcn_global_load_async_to_lds_b128)
#define HAVE_ASYNC_LDS 1
typedef __attribute__((address_space(1))) v4i* gv4i_p;   // global int4*
typedef __attribute__((address_space(3))) v4i* lv4i_p;   // LDS int4*
#else
#define HAVE_ASYNC_LDS 0
#endif

// 16B global -> LDS copy. Async (ASYNCcnt-tracked, no VGPR round trip) when
// the gfx1250 builtin exists; synchronous v4f copy otherwise.
__device__ __forceinline__ void cp16_to_lds(const float* __restrict__ g, float* l) {
#if HAVE_ASYNC_LDS
    __builtin_amdgcn_global_load_async_to_lds_b128((gv4i_p)g, (lv4i_p)l, 0, 0);
#else
    *(v4f*)l = *(const v4f*)g;
#endif
}

template <int N>
__device__ __forceinline__ void wait_async() {
#if HAVE_ASYNC_LDS
#if __has_builtin(__builtin_amdgcn_s_wait_asynccnt)
    __builtin_amdgcn_s_wait_asynccnt((unsigned short)N);
#else
    asm volatile("s_wait_asynccnt %0" :: "i"(N) : "memory");
#endif
#endif
}

// ---------------- K0: zero the bin accumulators in d_ws ----------------
__global__ void k0_zero(float* __restrict__ bins) {
    int i = blockIdx.x * 256 + threadIdx.x;
    if (i < 3 * BPN) bins[i] = 0.0f;
}

// ---------------- K1: LDS-privatized binning (bandwidth-bound phase) ----
// 2048 blocks x 256 threads, 32 points/thread. 3KB LDS bins per block,
// single global float-atomic flush per bin.
__global__ __launch_bounds__(256) void k1_bin(const float* __restrict__ pers,
                                              const int*   __restrict__ pix,
                                              float* __restrict__ bins) {
    __shared__ float sb[PP], sp[PP], sc[PP];
    const int t = threadIdx.x;
    sb[t] = 0.0f; sp[t] = 0.0f; sc[t] = 0.0f;
    __syncthreads();

    const int blocksPerBatch = NN / 8192;              // 32
    const int b     = blockIdx.x / blocksPerBatch;
    const int chunk = blockIdx.x % blocksPerBatch;
    const long long base = (long long)b * NN + (long long)chunk * 8192;

    for (int it = 0; it < 32; ++it) {
        long long i = base + it * 256 + t;
        int2 pc = ((const int2*)pix)[i];               // coalesced b64
        const float* c3 = pers + i * 3;                // coalesced 12B/pt
        float cx = c3[0], cy = c3[1], cz = c3[2];
        bool zero = (pc.x == 0 && pc.y == 0 &&
                     cx == 0.0f && cy == 0.0f && cz == 0.0f);
        if (!zero) {
            int px = pc.x / 14; if (px > 15) px = 15;  // matches fp32 div+trunc
            int py = pc.y / 14; if (py > 15) py = 15;
            int f = py * 16 + px;
            atomicAdd(&sb[f], cx);
            atomicAdd(&sp[f], cz);
            atomicAdd(&sc[f], 1.0f);
        }
    }
    __syncthreads();
    int g = b * PP + t;
    if (sc[t] != 0.0f) {
        atomicAdd(&bins[g],           sb[t]);
        atomicAdd(&bins[BPN + g],     sp[t]);
        atomicAdd(&bins[2 * BPN + g], sc[t]);
    }
}

// ---------------- K2: averages + scalar->64->64 MLPs + assemble combined ---
// combined row layout: [ spatial_pos(128) | birth_enc(64) | pers_enc(64) ]
// staged into d_out (overwritten later by K3).
__global__ __launch_bounds__(64) void k2_encode(
        const float* __restrict__ bins, const float* __restrict__ spatial,
        const float* __restrict__ bw1, const float* __restrict__ bb1,
        const float* __restrict__ bw2, const float* __restrict__ bb2,
        const float* __restrict__ pw1, const float* __restrict__ pb1,
        const float* __restrict__ pw2, const float* __restrict__ pb2,
        float* __restrict__ out) {
    const int row = blockIdx.x;            // b*P + p
    const int p   = row & (PP - 1);
    const int j   = threadIdx.x;
    __shared__ float zb[D4], zp[D4];

    float c = bins[2 * BPN + row];
    float bavg = 0.0f, pavg = 0.0f;
    if (c > 0.0f) { bavg = bins[row] / c; pavg = bins[BPN + row] / c; }

    float z1 = bavg * bw1[j] + bb1[j]; zb[j] = z1 > 0.0f ? z1 : 0.0f;
    float z2 = pavg * pw1[j] + pb1[j]; zp[j] = z2 > 0.0f ? z2 : 0.0f;
    __syncthreads();

    float eb = bb2[j], ep = pb2[j];
#pragma unroll 8
    for (int k = 0; k < D4; ++k) {
        eb += zb[k] * bw2[k * D4 + j];
        ep += zp[k] * pw2[k * D4 + j];
    }
    long long ro = (long long)row * DIM;
    out[ro + j]       = spatial[p * D2 + j];
    out[ro + 64 + j]  = spatial[p * D2 + 64 + j];
    out[ro + 128 + j] = eb;
    out[ro + 192 + j] = ep;
}

// ---------------- K3: WMMA GEMM (16384x256 @ 256x256) + LayerNorm + tanh ---
// Double-buffered async global->LDS staging overlapped with the WMMA K-loop.
#define ROWS      128
#define KC        32
#define NCHUNK    (DIM / KC)   // 8
#define SA_STRIDE 36           // 144B: b64-aligned, bank-spread
#define SF_STRIDE 260          // 1040B: 16B-aligned

__global__ __launch_bounds__(256) void k3_gemm_ln(
        const float* __restrict__ fw,  const float* __restrict__ fb,
        const float* __restrict__ lnw, const float* __restrict__ lnb,
        float* __restrict__ io) {
    __shared__ __align__(16) float sa[2][ROWS * SA_STRIDE];
    __shared__ __align__(16) float sf[2][KC * SF_STRIDE];

    const int tid  = threadIdx.x;
    const int wave = tid >> 5, lane = tid & 31;
    const int n = lane & 15, h = lane >> 4;           // wave32 WMMA lane split
    const int rbase = wave << 4;                      // this wave's 16-row M tile
    const long long rowbase = (long long)blockIdx.x * ROWS;

    v8f acc[16];
#pragma unroll
    for (int nt = 0; nt < 16; ++nt) acc[nt] = (v8f)0.0f;

    // Issue 12 copies per thread per chunk: 4 for A (128x32), 8 for fw (32x256).
    auto stage = [&](int kc, float* sab, float* sfb) {
#pragma unroll
        for (int i = 0; i < 4; ++i) {
            int idx = tid + i * 256;                  // 0..1023 16B chunks
            int r = idx >> 3, cg = (idx & 7) << 2;
            cp16_to_lds(io + (rowbase + r) * DIM + kc + cg, sab + r * SA_STRIDE + cg);
        }
#pragma unroll
        for (int i = 0; i < 8; ++i) {
            int idx = tid + i * 256;                  // 0..2047 16B chunks
            int r = idx >> 6, cg = (idx & 63) << 2;
            cp16_to_lds(fw + (kc + r) * DIM + cg, sfb + r * SF_STRIDE + cg);
        }
    };

    stage(0, sa[0], sf[0]);                           // preload chunk 0

    for (int c = 0; c < NCHUNK; ++c) {
        const int cur = c & 1;
        if (c + 1 < NCHUNK) {
            __syncthreads();                          // buf cur^1 free (compute of c-1 done)
            stage((c + 1) * KC, sa[cur ^ 1], sf[cur ^ 1]);   // prefetch next chunk
            wait_async<12>();                         // chunk c complete (in-order), next in flight
        } else {
            wait_async<0>();
        }
        __syncthreads();                              // publish chunk c to all waves

        const float* sab = sa[cur];
        const float* sfb = sf[cur];
#if __has_builtin(__builtin_amdgcn_wmma_f32_16x16x4_f32)
        // Full-precision fp32 WMMA: 8 K-steps of 4 per chunk.
        // A 16x4 layout: lane m=lane&15, a.x=A[m][k+2h], a.y=A[m][k+2h+1]
#pragma unroll
        for (int ks = 0; ks < KC; ks += 4) {
            int k0 = ks + (h << 1);
            v2f a = *(const v2f*)(sab + (rbase + n) * SA_STRIDE + k0);
#pragma unroll
            for (int nt = 0; nt < 16; ++nt) {
                v2f bf;
                bf.x = sfb[k0 * SF_STRIDE + (nt << 4) + n];
                bf.y = sfb[(k0 + 1) * SF_STRIDE + (nt << 4) + n];
                acc[nt] = __builtin_amdgcn_wmma_f32_16x16x4_f32(
                    false, a, false, bf, (short)0, acc[nt], false, false);
            }
        }
#else
        // Fallback: codegen-confirmed f16 WMMA, one 16x16x32 per N tile.
        v16h a16;
#pragma unroll
        for (int i = 0; i < 16; ++i) {
            int g = i >> 3, jj = i & 7;
            a16[i] = (_Float16)sab[(rbase + n) * SA_STRIDE + (h << 4) + i];
        }
#pragma unroll
        for (int nt = 0; nt < 16; ++nt) {
            v16h b16;
#pragma unroll
            for (int i = 0; i < 16; ++i)
                b16[i] = (_Float16)sfb[((h << 4) + i) * SF_STRIDE + (nt << 4) + n];
            acc[nt] = __builtin_amdgcn_wmma_f32_16x16x32_f16(
                false, a16, false, b16, (short)0, acc[nt], false, false);
        }
#endif
    }

    // + fb, then per-row LayerNorm in-register.
#pragma unroll
    for (int nt = 0; nt < 16; ++nt) {
        float fbv = fb[(nt << 4) + n];
#pragma unroll
        for (int r = 0; r < 8; ++r) acc[nt][r] += fbv;
    }
    // C/D layout: VGPR r -> row r (lanes 0-15) / row r+8 (lanes 16-31).
    float mu[8], rstd[8];
#pragma unroll
    for (int r = 0; r < 8; ++r) {
        float s = 0.0f, ss = 0.0f;
#pragma unroll
        for (int nt = 0; nt < 16; ++nt) { float v = acc[nt][r]; s += v; ss += v * v; }
#pragma unroll
        for (int m = 1; m < 16; m <<= 1) {            // butterfly within 16-lane half
            s  += __shfl_xor(s, m, 32);
            ss += __shfl_xor(ss, m, 32);
        }
        float mean = s * (1.0f / 256.0f);
        float var  = ss * (1.0f / 256.0f) - mean * mean;
        mu[r] = mean; rstd[r] = 1.0f / sqrtf(var + 1e-5f);
    }
    const long long outrow0 = rowbase + rbase + (h << 3);
#pragma unroll
    for (int nt = 0; nt < 16; ++nt) {
        int col = (nt << 4) + n;
        float w = lnw[col], bi = lnb[col];
#pragma unroll
        for (int r = 0; r < 8; ++r) {
            float v = tanhf((acc[nt][r] - mu[r]) * rstd[r] * w + bi);
            io[(outrow0 + r) * DIM + col] = v;
        }
    }
}

extern "C" void kernel_launch(void* const* d_in, const int* in_sizes, int n_in,
                              void* d_out, int out_size, void* d_ws, size_t ws_size,
                              hipStream_t stream) {
    const float* pers    = (const float*)d_in[1];
    const int*   pix     = (const int*)  d_in[2];
    const float* spatial = (const float*)d_in[3];
    const float* bw1 = (const float*)d_in[4];  const float* bb1 = (const float*)d_in[5];
    const float* bw2 = (const float*)d_in[6];  const float* bb2 = (const float*)d_in[7];
    const float* pw1 = (const float*)d_in[8];  const float* pb1 = (const float*)d_in[9];
    const float* pw2 = (const float*)d_in[10]; const float* pb2 = (const float*)d_in[11];
    const float* fw  = (const float*)d_in[12]; const float* fb  = (const float*)d_in[13];
    const float* lnw = (const float*)d_in[14]; const float* lnb = (const float*)d_in[15];
    float* out  = (float*)d_out;
    float* bins = (float*)d_ws;                        // 3*16384 floats = 192KB

    k0_zero  <<<(3 * BPN + 255) / 256, 256, 0, stream>>>(bins);
    k1_bin   <<<(BB * NN) / 8192,      256, 0, stream>>>(pers, pix, bins);
    k2_encode<<<BPN,                    64, 0, stream>>>(bins, spatial,
                 bw1, bb1, bw2, bb2, pw1, pb1, pw2, pb2, out);
    k3_gemm_ln<<<BPN / ROWS,           256, 0, stream>>>(fw, fb, lnw, lnb, out);
}
